// Transformer_9809705304095
// MI455X (gfx1250) — matrix operand order repaired and checked by reading, not compiled, not run on hardware
//
#include <hip/hip_runtime.h>
#include <hip/hip_bf16.h>
#include <math.h>

// ---------------------------------------------------------------------------
// Transformer forward (encoder-decoder) for gfx1250.
// GEMMs: v_wmma_f32_16x16x32_f16, 128x{128|64} block tiles, 8 wave32 waves,
// A-tile staged global->LDS via async-load path (ASYNCcnt), B-tile staged
// through registers with f32->f16 conversion into [n][k] fragment layout.
// ---------------------------------------------------------------------------

typedef __attribute__((ext_vector_type(16))) _Float16 v16h;
typedef __attribute__((ext_vector_type(8)))  float    v8f;
typedef __attribute__((ext_vector_type(4)))  int      v4i;

static constexpr int kD  = 512;     // d_model
static constexpr int kH  = 8;       // heads
static constexpr int kL  = 6;       // layers
static constexpr int kFF = 2048;    // d_ff
static constexpr int kS  = 512;     // seq
static constexpr int kB  = 8;       // batch
static constexpr int kV  = 32000;   // vocab
static constexpr int kDh = kD / kH; // 64
static constexpr int kM  = kB * kS; // 4096 tokens

// ---- CDNA5 async global->LDS copy (16B per lane) --------------------------
__device__ __forceinline__ void async_copy16(const void* g, void* l)
{
#if __has_builtin(__builtin_amdgcn_global_load_async_to_lds_b128)
    __builtin_amdgcn_global_load_async_to_lds_b128(
        (__attribute__((address_space(1))) v4i*)g,
        (__attribute__((address_space(3))) v4i*)l, 0, 0);
#else
    unsigned lofs = (unsigned)(unsigned long long)l;   // LDS offset = low 32 bits
    asm volatile("global_load_async_to_lds_b128 %0, %1, off"
                 :: "v"(lofs), "v"(g) : "memory");
#endif
}

__device__ __forceinline__ void wait_async0()
{
#if __has_builtin(__builtin_amdgcn_s_wait_asynccnt)
    __builtin_amdgcn_s_wait_asynccnt(0);
#else
    asm volatile("s_wait_asynccnt 0x0" ::: "memory");
#endif
}

// ---------------------------------------------------------------------------
// Tiled WMMA GEMM:  C[z] = alpha * A[z](MxK) x B[z](KxN) (+bias) (ReLU?)
// Block: 256 threads = 8 waves; C tile 128 x TN (TN = 128 or 64).
// Wave grid 4x2; wave tile 32 x (TN/2) as 2 x NF wmma fragments.
// TRANSB==1: B stored N x K row-major (Q*K^T). M % 128 == 0, K % 32 == 0.
// z batching: operand offset = (z/zh)*s1 + (z%zh)*s2  (covers (b,h) loops).
// ---------------------------------------------------------------------------
template<int TN, int TRANSB, int RELU, int HASBIAS>
__global__ __launch_bounds__(256)
void gemm_wmma_kernel(const float* __restrict__ A, const float* __restrict__ Bm,
                      const float* __restrict__ bias, float* __restrict__ C,
                      int M, int N, int K, int lda, int ldb, int ldc,
                      int zh, long sA1, long sA2, long sB1, long sB2,
                      long sC1, long sC2, float alpha)
{
    constexpr int WC = TN / 2;     // wave col width (64 or 32)
    constexpr int NF = WC / 16;    // B fragments per wave (4 or 2)

    __shared__ float    As[128][36];   // fp32 A tile (async-loaded), padded
    __shared__ _Float16 Bs[TN][40];    // f16 B tile [n][k], padded

    const int z = blockIdx.z;
    const float* Ab = A  + (long)(z / zh) * sA1 + (long)(z % zh) * sA2;
    const float* Bb = Bm + (long)(z / zh) * sB1 + (long)(z % zh) * sB2;
    float*       Cb = C  + (long)(z / zh) * sC1 + (long)(z % zh) * sC2;

    const int tid  = threadIdx.x;
    const int wave = tid >> 5;
    const int lane = tid & 31;
    const int m0 = blockIdx.y * 128;
    const int n0 = blockIdx.x * TN;
    const int wm = (wave & 3) * 32;    // wave row offset in tile
    const int wn = (wave >> 2) * WC;   // wave col offset in tile

    v8f acc[2][NF] = {};

    for (int k0 = 0; k0 < K; k0 += 32) {
        // ---- A tile: 128 rows x 32 f32, DMA'd as 16B chunks (1024 chunks) ----
#pragma unroll
        for (int i = 0; i < 4; ++i) {
            int chunk = tid + 256 * i;
            int r = chunk >> 3, seg = chunk & 7;
            async_copy16(Ab + (long)(m0 + r) * lda + (k0 + seg * 4),
                         &As[r][seg * 4]);
        }
        // ---- B tile -> [n][k] f16 via register convert ----
        if (TRANSB) {
            for (int i = 0; i < TN / 8; ++i) {
                int idx = tid + 256 * i;
                int r = idx >> 5, c = idx & 31;        // r = n, c = k
                Bs[r][c] = (_Float16)Bb[(long)(n0 + r) * ldb + (k0 + c)];
            }
        } else {
            for (int i = 0; i < TN / 8; ++i) {
                int idx = tid + 256 * i;
                int c = idx / TN, n = idx % TN;        // c = k (0..31)
                Bs[n][c] = (_Float16)Bb[(long)(k0 + c) * ldb + (n0 + n)];
            }
        }
        wait_async0();        // this wave's async LDS writes complete
        __syncthreads();      // all waves' tiles visible

        // ---- fragment fills: row/col = lane%16, contiguous k = 16*lgrp + i in both A and B ----
        const int lgrp = lane >> 4;    // 0/1
        const int lr   = lane & 15;
        const int akb  = lgrp * 16;    // A: k-base 0 or 16
        const int bkb  = lgrp * 16;    // B: k-base 0 or 16

        v16h af[2];
        v16h bf[NF];
#pragma unroll
        for (int mi = 0; mi < 2; ++mi) {
            const int row = wm + 16 * mi + lr;
#pragma unroll
            for (int i = 0; i < 16; ++i) {
                int kk = akb + i;  // contiguous k
                af[mi][i] = (_Float16)As[row][kk];
            }
        }
#pragma unroll
        for (int ni = 0; ni < NF; ++ni) {
            const int col = wn + 16 * ni + lr;
#pragma unroll
            for (int i = 0; i < 16; ++i)
                bf[ni][i] = Bs[col][bkb + i];
        }

#pragma unroll
        for (int mi = 0; mi < 2; ++mi)
#pragma unroll
            for (int ni = 0; ni < NF; ++ni)
                acc[mi][ni] = __builtin_amdgcn_wmma_f32_16x16x32_f16(
                    false, af[mi], false, bf[ni], (short)0, acc[mi][ni],
                    false, false);
        __syncthreads();
    }

    // ---- store: C/D layout: lane 0-15 -> N=lane, M=j; lane 16-31 -> M=8+j ----
    const int lr = lane & 15;
    const int mb = (lane >> 4) * 8;
#pragma unroll
    for (int mi = 0; mi < 2; ++mi)
#pragma unroll
        for (int ni = 0; ni < NF; ++ni) {
            const int gn = n0 + wn + 16 * ni + lr;
            const float bv = HASBIAS ? bias[gn] : 0.0f;
#pragma unroll
            for (int j = 0; j < 8; ++j) {
                const int gm = m0 + wm + 16 * mi + mb + j;
                float val = alpha * acc[mi][ni][j] + bv;
                if (RELU) val = val > 0.0f ? val : 0.0f;
                Cb[(long)gm * ldc + gn] = val;
            }
        }
}

// ---------------------------------------------------------------------------
// Embedding * sqrt(D) + sinusoidal positional encoding (computed on the fly)
// ---------------------------------------------------------------------------
__global__ void embed_kernel(const int* __restrict__ tok,
                             const float* __restrict__ emb,
                             float* __restrict__ out, int S, int D, int total)
{
    int idx = blockIdx.x * blockDim.x + threadIdx.x;
    if (idx >= total) return;
    int d = idx % D;
    int s = (idx / D) % S;
    int t = tok[idx / D];
    float scale = sqrtf((float)D);
    int i2 = d & ~1;
    float div = __expf((float)i2 * (-logf(10000.0f) / (float)D));
    float ang = (float)s * div;
    float pe = (d & 1) ? __cosf(ang) : __sinf(ang);
    out[idx] = emb[(long)t * D + d] * scale + pe;
}

// ---------------------------------------------------------------------------
// Masked softmax over last dim (S=512), one block of 128 threads per row.
// mode 0: key padding mask  valid = (tok[b,k] != 0)            (enc / cross)
// mode 1: query padding + causal: valid = (tok[b,q]!=0) && k<=q (dec self)
// ---------------------------------------------------------------------------
__global__ __launch_bounds__(128)
void softmax_mask_kernel(float* __restrict__ scores, const int* __restrict__ tok,
                         int H, int S, int mode)
{
    const long row = blockIdx.x;          // (b*H + h)*S + q
    const int  q   = (int)(row % S);
    const long bh  = row / S;
    const int  b   = (int)(bh / H);
    float* p = scores + row * (long)S;
    const int tid = threadIdx.x;

    __shared__ float red[128];
    float vals[4];
    bool qvalid = true;
    if (mode == 1) qvalid = (tok[b * S + q] != 0);

    float mx = -1e30f;
#pragma unroll
    for (int i = 0; i < 4; ++i) {
        int kk = tid + 128 * i;
        float v = p[kk];
        bool valid = (mode == 0) ? (tok[b * S + kk] != 0)
                                 : (qvalid && (kk <= q));
        v = valid ? v : -1e9f;
        vals[i] = v;
        mx = fmaxf(mx, v);
    }
    red[tid] = mx; __syncthreads();
    for (int st = 64; st > 0; st >>= 1) {
        if (tid < st) red[tid] = fmaxf(red[tid], red[tid + st]);
        __syncthreads();
    }
    mx = red[0]; __syncthreads();

    float sum = 0.0f;
#pragma unroll
    for (int i = 0; i < 4; ++i) { vals[i] = __expf(vals[i] - mx); sum += vals[i]; }
    red[tid] = sum; __syncthreads();
    for (int st = 64; st > 0; st >>= 1) {
        if (tid < st) red[tid] += red[tid + st];
        __syncthreads();
    }
    const float inv = 1.0f / red[0];
#pragma unroll
    for (int i = 0; i < 4; ++i) p[tid + 128 * i] = vals[i] * inv;
}

// ---------------------------------------------------------------------------
// out = LayerNorm(a + r) * gs + gb     (D = 512, in-place over r is safe)
// ---------------------------------------------------------------------------
__global__ __launch_bounds__(128)
void layernorm_kernel(const float* __restrict__ a, const float* __restrict__ r,
                      const float* __restrict__ gs, const float* __restrict__ gb,
                      float* __restrict__ out, int D)
{
    const long row = blockIdx.x;
    const int tid = threadIdx.x;
    const float* pa = a + row * (long)D;
    const float* pr = r + row * (long)D;
    float* po = out + row * (long)D;

    __shared__ float red[128];
    float v[4];
    float sum = 0.0f;
#pragma unroll
    for (int i = 0; i < 4; ++i) {
        int d = tid + 128 * i;
        v[i] = pa[d] + pr[d];
        sum += v[i];
    }
    red[tid] = sum; __syncthreads();
    for (int st = 64; st > 0; st >>= 1) { if (tid < st) red[tid] += red[tid + st]; __syncthreads(); }
    const float mean = red[0] / (float)D; __syncthreads();

    float s2 = 0.0f;
#pragma unroll
    for (int i = 0; i < 4; ++i) { float dd = v[i] - mean; s2 += dd * dd; }
    red[tid] = s2; __syncthreads();
    for (int st = 64; st > 0; st >>= 1) { if (tid < st) red[tid] += red[tid + st]; __syncthreads(); }
    const float inv = rsqrtf(red[0] / (float)D + 1e-5f);
#pragma unroll
    for (int i = 0; i < 4; ++i) {
        int d = tid + 128 * i;
        po[d] = (v[i] - mean) * inv * gs[d] + gb[d];
    }
}

// ---------------------------------------------------------------------------
// Host-side helpers
// ---------------------------------------------------------------------------
static void launch_gemm(hipStream_t st, const float* A, const float* B, const float* bias,
                        float* C, int M, int N, int K, int lda, int ldb, int ldc,
                        bool transb, bool relu, float alpha,
                        int z, int zh, long sA1, long sA2, long sB1, long sB2,
                        long sC1, long sC2)
{
    dim3 blk(256);
    if (N % 128 == 0) {
        dim3 grid((unsigned)(N / 128), (unsigned)(M / 128), (unsigned)z);
        if (transb)
            gemm_wmma_kernel<128, 1, 0, 0><<<grid, blk, 0, st>>>(A, B, bias, C, M, N, K,
                lda, ldb, ldc, zh, sA1, sA2, sB1, sB2, sC1, sC2, alpha);
        else if (relu)
            gemm_wmma_kernel<128, 0, 1, 1><<<grid, blk, 0, st>>>(A, B, bias, C, M, N, K,
                lda, ldb, ldc, zh, sA1, sA2, sB1, sB2, sC1, sC2, alpha);
        else if (bias)
            gemm_wmma_kernel<128, 0, 0, 1><<<grid, blk, 0, st>>>(A, B, bias, C, M, N, K,
                lda, ldb, ldc, zh, sA1, sA2, sB1, sB2, sC1, sC2, alpha);
        else
            gemm_wmma_kernel<128, 0, 0, 0><<<grid, blk, 0, st>>>(A, B, bias, C, M, N, K,
                lda, ldb, ldc, zh, sA1, sA2, sB1, sB2, sC1, sC2, alpha);
    } else {  // N == 64 (attn x V)
        dim3 grid((unsigned)(N / 64), (unsigned)(M / 128), (unsigned)z);
        if (bias)
            gemm_wmma_kernel<64, 0, 0, 1><<<grid, blk, 0, st>>>(A, B, bias, C, M, N, K,
                lda, ldb, ldc, zh, sA1, sA2, sB1, sB2, sC1, sC2, alpha);
        else
            gemm_wmma_kernel<64, 0, 0, 0><<<grid, blk, 0, st>>>(A, B, bias, C, M, N, K,
                lda, ldb, ldc, zh, sA1, sA2, sB1, sB2, sC1, sC2, alpha);
    }
}

// Multi-head attention: out = MHA(qin attends kvin), W = 4 x [D,D], Wb = 4 x [D]
static void run_attention(hipStream_t st, const float* qin, const float* kvin,
                          const float* W, const float* Wb, const int* tok, int mode,
                          float* q, float* k, float* v, float* scores,
                          float* attn, float* out)
{
    const long DD = (long)kD * kD;
    launch_gemm(st, qin,  W + 0 * DD, Wb + 0 * kD, q, kM, kD, kD, kD, kD, kD,
                false, false, 1.f, 1, 1, 0, 0, 0, 0, 0, 0);
    launch_gemm(st, kvin, W + 1 * DD, Wb + 1 * kD, k, kM, kD, kD, kD, kD, kD,
                false, false, 1.f, 1, 1, 0, 0, 0, 0, 0, 0);
    launch_gemm(st, kvin, W + 2 * DD, Wb + 2 * kD, v, kM, kD, kD, kD, kD, kD,
                false, false, 1.f, 1, 1, 0, 0, 0, 0, 0, 0);
    // scores[b,h] = (Q_bh x K_bh^T) / sqrt(dh);  z = b*H + h
    launch_gemm(st, q, k, nullptr, scores, kS, kS, kDh, kD, kD, kS,
                true, false, 0.125f, kB * kH, kH,
                (long)kS * kD, (long)kDh,
                (long)kS * kD, (long)kDh,
                (long)kH * kS * kS, (long)kS * kS);
    softmax_mask_kernel<<<kB * kH * kS, 128, 0, st>>>(scores, tok, kH, kS, mode);
    // context[b,h] = attn x V_bh  -> written into [b, s, h*64 + d]
    launch_gemm(st, scores, v, nullptr, attn, kS, kDh, kS, kS, kD, kD,
                false, false, 1.f, kB * kH, kH,
                (long)kH * kS * kS, (long)kS * kS,
                (long)kS * kD, (long)kDh,
                (long)kS * kD, (long)kDh);
    launch_gemm(st, attn, W + 3 * DD, Wb + 3 * kD, out, kM, kD, kD, kD, kD, kD,
                false, false, 1.f, 1, 1, 0, 0, 0, 0, 0, 0);
}

extern "C" void kernel_launch(void* const* d_in, const int* in_sizes, int n_in,
                              void* d_out, int out_size, void* d_ws, size_t ws_size,
                              hipStream_t stream)
{
    (void)in_sizes; (void)n_in; (void)out_size; (void)ws_size;

    const int*   src        = (const int*)d_in[0];
    const int*   tgt        = (const int*)d_in[1];
    const float* src_emb    = (const float*)d_in[2];
    const float* tgt_emb    = (const float*)d_in[3];
    const float* enc_attn_w = (const float*)d_in[4];
    const float* enc_attn_b = (const float*)d_in[5];
    const float* enc_ln_s   = (const float*)d_in[6];
    const float* enc_ln_b   = (const float*)d_in[7];
    const float* enc_ff_w1  = (const float*)d_in[8];
    const float* enc_ff_b1  = (const float*)d_in[9];
    const float* enc_ff_w2  = (const float*)d_in[10];
    const float* enc_ff_b2  = (const float*)d_in[11];
    const float* dec_attn_w = (const float*)d_in[12];
    const float* dec_attn_b = (const float*)d_in[13];
    const float* dec_ln_s   = (const float*)d_in[14];
    const float* dec_ln_b   = (const float*)d_in[15];
    const float* dec_ff_w1  = (const float*)d_in[16];
    const float* dec_ff_b1  = (const float*)d_in[17];
    const float* dec_ff_w2  = (const float*)d_in[18];
    const float* dec_ff_b2  = (const float*)d_in[19];
    const float* fc_w       = (const float*)d_in[20];
    const float* fc_b       = (const float*)d_in[21];
    float* outp = (float*)d_out;

    // ---- carve workspace (~160 MB) ----
    float* base = (float*)d_ws;
    size_t off = 0;
    auto carve = [&](size_t n) { float* p = base + off; off += n; return p; };
    const size_t AN = (size_t)kM * kD;
    float* x      = carve(AN);
    float* y      = carve(AN);
    float* q      = carve(AN);
    float* k      = carve(AN);
    float* v      = carve(AN);
    float* attn   = carve(AN);
    float* a1     = carve(AN);
    float* ffh    = carve((size_t)kM * kFF);
    float* scores = carve((size_t)kB * kH * kS * kS);

    // ---- embeddings + positional encoding ----
    const int tot = kM * kD;
    embed_kernel<<<(tot + 255) / 256, 256, 0, stream>>>(src, src_emb, x, kS, kD, tot);
    embed_kernel<<<(tot + 255) / 256, 256, 0, stream>>>(tgt, tgt_emb, y, kS, kD, tot);

    const long DD = (long)kD * kD;

    // ---- encoder ----
    for (int i = 0; i < kL; ++i) {
        run_attention(stream, x, x, enc_attn_w + (long)i * 4 * DD,
                      enc_attn_b + (long)i * 4 * kD, src, /*mode*/0,
                      q, k, v, scores, attn, a1);
        layernorm_kernel<<<kM, 128, 0, stream>>>(a1, x,
            enc_ln_s + ((long)i * 2 + 0) * kD, enc_ln_b + ((long)i * 2 + 0) * kD, x, kD);
        launch_gemm(stream, x, enc_ff_w1 + (long)i * kD * kFF, enc_ff_b1 + (long)i * kFF,
                    ffh, kM, kFF, kD, kD, kFF, kFF, false, /*relu*/true, 1.f,
                    1, 1, 0, 0, 0, 0, 0, 0);
        launch_gemm(stream, ffh, enc_ff_w2 + (long)i * kFF * kD, enc_ff_b2 + (long)i * kD,
                    a1, kM, kD, kFF, kFF, kD, kD, false, false, 1.f,
                    1, 1, 0, 0, 0, 0, 0, 0);
        layernorm_kernel<<<kM, 128, 0, stream>>>(a1, x,
            enc_ln_s + ((long)i * 2 + 1) * kD, enc_ln_b + ((long)i * 2 + 1) * kD, x, kD);
    }

    // ---- decoder ----
    for (int i = 0; i < kL; ++i) {
        run_attention(stream, y, y, dec_attn_w + (long)i * 8 * DD,
                      dec_attn_b + (long)i * 8 * kD, tgt, /*mode*/1,
                      q, k, v, scores, attn, a1);
        layernorm_kernel<<<kM, 128, 0, stream>>>(a1, y,
            dec_ln_s + ((long)i * 3 + 0) * kD, dec_ln_b + ((long)i * 3 + 0) * kD, y, kD);
        run_attention(stream, y, x, dec_attn_w + (long)i * 8 * DD + 4 * DD,
                      dec_attn_b + (long)i * 8 * kD + 4 * kD, src, /*mode*/0,
                      q, k, v, scores, attn, a1);
        layernorm_kernel<<<kM, 128, 0, stream>>>(a1, y,
            dec_ln_s + ((long)i * 3 + 1) * kD, dec_ln_b + ((long)i * 3 + 1) * kD, y, kD);
        launch_gemm(stream, y, dec_ff_w1 + (long)i * kD * kFF, dec_ff_b1 + (long)i * kFF,
                    ffh, kM, kFF, kD, kD, kFF, kFF, false, true, 1.f,
                    1, 1, 0, 0, 0, 0, 0, 0);
        launch_gemm(stream, ffh, dec_ff_w2 + (long)i * kFF * kD, dec_ff_b2 + (long)i * kD,
                    a1, kM, kD, kFF, kFF, kD, kD, false, false, 1.f,
                    1, 1, 0, 0, 0, 0, 0, 0);
        layernorm_kernel<<<kM, 128, 0, stream>>>(a1, y,
            dec_ln_s + ((long)i * 3 + 2) * kD, dec_ln_b + ((long)i * 3 + 2) * kD, y, kD);
    }

    // ---- final projection to vocab ----
    launch_gemm(stream, y, fc_w, fc_b, outp, kM, kV, kD, kD, kV, kV,
                false, false, 1.f, 1, 1, 0, 0, 0, 0, 0, 0);
}
